// New_RFBMultiHAttnNetwork_V4_21311627722939
// MI455X (gfx1250) — compile-verified, matrix operand order checked by reading, not run on hardware
//
#include <hip/hip_runtime.h>
#include <hip/hip_bf16.h>

// ---------------------------------------------------------------------------
// Channel-attention net on gfx1250: BF16 WMMA (f32 accum) + TDM async staging.
//   XqT/XcT = transpose(query/context) -> bf16 [N,C]        (one-time)
//   Q = Wq XqT + bq  [F,N] bf16 ;  K = Wk XcT + bk  [F,N] bf16
//   VT = (Wv XcT + bv)^T            [N,F] bf16 (transposed store)
//   S = Q K^T (f32) ; PT = softmax(S)^T (bf16)
//   WP = Wo P  (bf16)   [fusion: Wo (P V) = (Wo P) V]
//   Y  = WP V + bo -> f32 d_out
// Every GEMM uses the same kernel: B-tile rows are output columns with
// K-contiguous bf16 -> staged by TENSOR_LOAD_TO_LDS (pad 16B per 64B row ->
// 80B LDS stride), A staged via register-pipelined vector loads (or TDM when
// A is bf16). Double-buffered LDS, one barrier per K-step, 8 wmma/wave/step.
// ---------------------------------------------------------------------------

typedef unsigned short u16;
typedef unsigned int   u32;
typedef __attribute__((ext_vector_type(4)))  unsigned short us4;
typedef __attribute__((ext_vector_type(8)))  unsigned short us8;
typedef __attribute__((ext_vector_type(16))) unsigned short us16;
typedef __attribute__((ext_vector_type(4)))  float f32x4;
typedef __attribute__((ext_vector_type(8)))  float v8f;
typedef __attribute__((ext_vector_type(16))) __bf16 v16bf;
typedef __attribute__((ext_vector_type(4)))  unsigned int u32x4;
typedef __attribute__((ext_vector_type(8)))  int i32x8;
typedef __attribute__((ext_vector_type(4)))  int i32x4;

__device__ __forceinline__ u16 f2bf(float x) {
    unsigned u = __builtin_bit_cast(unsigned, x);
    u += 0x7FFFu + ((u >> 16) & 1u);              // round to nearest even
    return (u16)(u >> 16);
}

__device__ __forceinline__ us4 ld4cvt(const float* __restrict__ s) {
    f32x4 f = *(const f32x4*)s;                   // global_load_b128
    us4 r = { f2bf(f[0]), f2bf(f[1]), f2bf(f[2]), f2bf(f[3]) };
    return r;
}
__device__ __forceinline__ us4 ld4cvt(const u16* __restrict__ s) {
    return *(const us4*)s;                        // global_load_b64
}

__device__ __forceinline__ void storeOut(float* p, float v) { *p = v; }
__device__ __forceinline__ void storeOut(u16* p, float v)   { *p = f2bf(v); }

// WMMA fragment from LDS tile stored [row][40 u16 (80B) stride]:
// lanes 0-15 -> row=lane, K 0..7 & 16..23 ; lanes 16-31 -> row=lane-16, K 8..15 & 24..31
__device__ __forceinline__ v16bf load_frag(const u16* __restrict__ base, int lane) {
    const u16* p = base + (lane & 15) * 40;
    const int k0 = (lane >> 4) << 3;
    union { us16 v; us8 h[2]; } f;
    f.h[0] = *(const us8*)(p + k0);               // ds_load_b128
    f.h[1] = *(const us8*)(p + 16 + k0);          // ds_load_b128
    return __builtin_bit_cast(v16bf, f.v);
}

// ---------------------------------------------------------------------------
// TDM: async-copy a [tileRows x 32] bf16 tile (row stride kTotal elements in
// global) into LDS with 16B pad after each 64B row -> 80B LDS row stride.
// D# per CDNA5 ISA ch.8: group0 = {count/lds_addr/global_addr/type},
// group1 = {data_size=2B, pad_enable, pad_interval=16DW, pad_amount=4DW,
//           tensor_dim0=kTotal, tensor_dim1=1M, tile=32 x tileRows,
//           dim0_stride=kTotal}. Groups 2/3 zero (2-D tensor).
// Toolchain uses the 6-arg builtin form: (g0, g1, g2, g3, g4, cpol).
// ---------------------------------------------------------------------------
__device__ __forceinline__ void tdm_tile_to_lds(const void* gptr, u32 ldsOff,
                                                int kTotal, int tileRows)
{
    const unsigned long long ga = (unsigned long long)(uintptr_t)gptr;
    u32x4 g0;
    g0[0] = 1u;                                              // count=1, no gather
    g0[1] = ldsOff;                                          // lds_addr
    g0[2] = (u32)ga;                                         // global_addr[31:0]
    g0[3] = ((u32)(ga >> 32) & 0x01FFFFFFu) | 0x80000000u;   // ga[56:32] | type=2
    i32x8 g1;
    g1[0] = (int)0x06D10000u;   // data_size=2B | pad_en | pad_int=16DW | pad_amt=4DW
    g1[1] = (int)(((u32)kTotal & 0xFFFFu) << 16);            // tensor_dim0 lo
    g1[2] = (int)(((u32)kTotal >> 16) & 0xFFFFu);            // dim0 hi | dim1 lo (=0)
    g1[3] = (int)(16u | (32u << 16));                        // dim1 hi (1M) | tile_dim0=32
    g1[4] = (int)((u32)tileRows & 0xFFFFu);                  // tile_dim1 | tile_dim2=0
    g1[5] = (int)(u32)kTotal;                                // tensor_dim0_stride lo
    g1[6] = 0;                                               // stride hi | dim1_stride lo
    g1[7] = 0;
    const i32x4 gz4 = { 0, 0, 0, 0 };
    const i32x8 gz8 = { 0, 0, 0, 0, 0, 0, 0, 0 };
    __builtin_amdgcn_tensor_load_to_lds(g0, g1, gz4, gz4, gz8, 0);
}

// ---------------------------------------------------------------------------
// GEMM: D[M,Nc] (+bias, opt. transposed store) = A[M,K] x B, batched over z.
//   A: [M,K] row-major (float -> cvt, or bf16 -> TDM-staged).
//   B: [Nc,K] row-major bf16 (rows are output columns), TDM-staged.
// WG: 256 thr (8 wave32), tile 256x64, K-step 32, per wave 64x32 = 8 WMMA.
// Double-buffered LDS; wave0 issues TDM one step ahead; A (float case)
// register-pipelined one step ahead.
// ---------------------------------------------------------------------------
template <typename TA, typename TOUT, bool STORE_T, bool HAS_BIAS>
__global__ __launch_bounds__(256)
void gemm_bt_wmma(const TA* __restrict__ Aall, const u16* __restrict__ Ball,
                  TOUT* __restrict__ Dall, const float* __restrict__ bias,
                  int M, int Nc, int K,
                  long long sA, long long sB, long long sD)
{
    constexpr int LK = 40;                         // 80B row stride (TDM pad)
    constexpr bool A_TDM = __is_same(TA, u16);
    __shared__ u16 ldsA[2][256 * LK];
    __shared__ u16 ldsB[2][64 * LK];

    const int tid  = threadIdx.x;
    const int lane = tid & 31;
    const int wave = tid >> 5;
    const int wm   = wave & 3;                     // 4 row groups x 64
    const int wn   = wave >> 2;                    // 2 col groups x 32
    const int rowBase = blockIdx.y * 256;
    const int colBase = blockIdx.x * 64;

    const TA*  A  = Aall + (size_t)blockIdx.z * sA;
    const u16* Bp = Ball + (size_t)blockIdx.z * sB;
    TOUT*      D  = Dall + (size_t)blockIdx.z * sD;

    const u32 ldsAoff[2] = { (u32)(uintptr_t)&ldsA[0][0], (u32)(uintptr_t)&ldsA[1][0] };
    const u32 ldsBoff[2] = { (u32)(uintptr_t)&ldsB[0][0], (u32)(uintptr_t)&ldsB[1][0] };

    const int r0 = tid >> 3;                       // 0..31
    const int kq = (tid & 7) * 4;                  // 0,4,..,28
    const TA* aSrc = A + (size_t)(rowBase + r0) * K + kq;

    // ---- prologue: stage K-step 0 into buffer 0
    us4 aR[8];
    if (A_TDM) {
        if (wave == 0) tdm_tile_to_lds(A + (size_t)rowBase * K, ldsAoff[0], K, 256);
    } else {
        #pragma unroll
        for (int it = 0; it < 8; ++it) aR[it] = ld4cvt(aSrc + (size_t)it * 32 * K);
    }
    if (wave == 0) tdm_tile_to_lds(Bp + (size_t)colBase * K, ldsBoff[0], K, 64);

    v8f acc[4][2] = {};
    int buf = 0;

    for (int kk = 0; kk < K; kk += 32) {
        if (!A_TDM) {
            #pragma unroll
            for (int it = 0; it < 8; ++it)
                *(us4*)&ldsA[buf][(it * 32 + r0) * LK + kq] = aR[it];
        }
        if (wave == 0) __builtin_amdgcn_s_wait_tensorcnt(0);
        __syncthreads();

        if (kk + 32 < K) {                         // stage next step into other buffer
            if (A_TDM) {
                if (wave == 0)
                    tdm_tile_to_lds(A + (size_t)rowBase * K + kk + 32, ldsAoff[buf ^ 1], K, 256);
            } else {
                #pragma unroll
                for (int it = 0; it < 8; ++it)
                    aR[it] = ld4cvt(aSrc + (size_t)it * 32 * K + kk + 32);
            }
            if (wave == 0)
                tdm_tile_to_lds(Bp + (size_t)colBase * K + kk + 32, ldsBoff[buf ^ 1], K, 64);
        }

        v16bf aF[4], bF[2];
        #pragma unroll
        for (int fm = 0; fm < 4; ++fm)
            aF[fm] = load_frag(&ldsA[buf][(wm * 64 + fm * 16) * LK], lane);
        #pragma unroll
        for (int fn = 0; fn < 2; ++fn)
            bF[fn] = load_frag(&ldsB[buf][(wn * 32 + fn * 16) * LK], lane);
        #pragma unroll
        for (int fm = 0; fm < 4; ++fm)
            #pragma unroll
            for (int fn = 0; fn < 2; ++fn)
                acc[fm][fn] = __builtin_amdgcn_wmma_f32_16x16x32_bf16(
                    false, aF[fm], false, bF[fn], (short)0, acc[fm][fn], false, false);
        buf ^= 1;
    }

    // ---- writeback (C/D layout: lane<16 -> M=j, lane>=16 -> M=8+j)
    const int colL = lane & 15;
    const int rOff = (lane >> 4) << 3;
    #pragma unroll
    for (int fm = 0; fm < 4; ++fm) {
        const int rowT = rowBase + wm * 64 + fm * 16 + rOff;
        #pragma unroll
        for (int fn = 0; fn < 2; ++fn) {
            const int col = colBase + wn * 32 + fn * 16 + colL;
            #pragma unroll
            for (int j = 0; j < 8; ++j) {
                const int row = rowT + j;
                float v = acc[fm][fn][j];
                if (HAS_BIAS) v += bias[row];
                if (STORE_T) storeOut(&D[(size_t)col * M + row], v);   // D^T [Nc,M]
                else         storeOut(&D[(size_t)row * Nc + col], v);
            }
        }
    }
}

// ---------------------------------------------------------------------------
// Transpose+convert: X [C,N] f32 -> XT [N,C] bf16 (64x64 tiles via LDS).
// ---------------------------------------------------------------------------
__global__ __launch_bounds__(256)
void transpose_f32_bf16(const float* __restrict__ X, u16* __restrict__ XT,
                        int C, int N)
{
    __shared__ u16 tile[64][68];
    const int n0 = blockIdx.x * 64;
    const int c0 = blockIdx.y * 64;
    const float* Xb  = X  + (size_t)blockIdx.z * C * N;
    u16*         XTb = XT + (size_t)blockIdx.z * C * N;

    const int tn = threadIdx.x & 63;
    const int tc = threadIdx.x >> 6;               // 0..3
    #pragma unroll
    for (int it = 0; it < 16; ++it) {
        const int c = it * 4 + tc;
        tile[c][tn] = f2bf(Xb[(size_t)(c0 + c) * N + n0 + tn]);
    }
    __syncthreads();
    const int cq = (threadIdx.x & 15) * 4;
    const int n1 = threadIdx.x >> 4;               // 0..15
    #pragma unroll
    for (int it = 0; it < 4; ++it) {
        const int n = it * 16 + n1;
        us4 v = { tile[cq][n], tile[cq + 1][n], tile[cq + 2][n], tile[cq + 3][n] };
        *(us4*)&XTb[(size_t)(n0 + n) * C + c0 + cq] = v;
    }
}

// ---------------------------------------------------------------------------
// Row softmax (512 wide) -> transposed bf16 output PT[g,f] (32 f-rows/block,
// LDS-transposed so stores are us4-packed along f).
// ---------------------------------------------------------------------------
__global__ __launch_bounds__(256)
void softmax512_to_colmajor(const float* __restrict__ S, u16* __restrict__ PT)
{
    __shared__ u16 tile[32][520];
    const int lane = threadIdx.x & 31;
    const int wave = threadIdx.x >> 5;
    const size_t base = (size_t)blockIdx.z * 512 * 512;

    #pragma unroll
    for (int j = 0; j < 4; ++j) {
        const int fl = wave * 4 + j;
        const float* s = S + base + (size_t)(blockIdx.x * 32 + fl) * 512;
        float v[16];
        float m = -3.402823466e38f;
        #pragma unroll
        for (int i = 0; i < 16; ++i) { v[i] = s[lane + i * 32]; m = fmaxf(m, v[i]); }
        #pragma unroll
        for (int off = 16; off >= 1; off >>= 1) m = fmaxf(m, __shfl_xor(m, off, 32));
        float sum = 0.f;
        #pragma unroll
        for (int i = 0; i < 16; ++i) { v[i] = __expf(v[i] - m); sum += v[i]; }
        #pragma unroll
        for (int off = 16; off >= 1; off >>= 1) sum += __shfl_xor(sum, off, 32);
        const float inv = 1.f / sum;
        #pragma unroll
        for (int i = 0; i < 16; ++i) tile[fl][lane + i * 32] = f2bf(v[i] * inv);
    }
    __syncthreads();

    const int fq = (threadIdx.x & 7) * 4;
    const int g0 = threadIdx.x >> 3;               // 0..31
    u16* out = PT + base + blockIdx.x * 32 + fq;
    #pragma unroll
    for (int it = 0; it < 16; ++it) {
        const int g = it * 32 + g0;
        us4 w = { tile[fq][g], tile[fq + 1][g], tile[fq + 2][g], tile[fq + 3][g] };
        *(us4*)&out[(size_t)g * 512] = w;
    }
}

// ---------------------------------------------------------------------------
extern "C" void kernel_launch(void* const* d_in, const int* in_sizes, int n_in,
                              void* d_out, int out_size, void* d_ws, size_t ws_size,
                              hipStream_t stream) {
    constexpr int Bn = 32, C = 512, F = 512, N = 56 * 56;   // N = 3136

    const float* query   = (const float*)d_in[0];
    const float* context = (const float*)d_in[1];
    const float* Wq = (const float*)d_in[2];
    const float* bq = (const float*)d_in[3];
    const float* Wk = (const float*)d_in[4];
    const float* bk = (const float*)d_in[5];
    const float* Wv = (const float*)d_in[6];
    const float* bv = (const float*)d_in[7];
    const float* Wo = (const float*)d_in[8];
    const float* bo = (const float*)d_in[9];

    // ws: XqT|XcT|Qb|Kb|VbT (bf16, B*F*N each) | S (f32 B*F*F) | PbT | WPb (bf16)
    const size_t eT = (size_t)Bn * F * N;
    const size_t eS = (size_t)Bn * F * F;
    u16*   XqT = (u16*)d_ws;
    u16*   XcT = XqT + eT;
    u16*   Qb  = XcT + eT;
    u16*   Kb  = Qb  + eT;
    u16*   VbT = Kb  + eT;
    float* S   = (float*)(VbT + eT);
    u16*   PbT = (u16*)(S + eS);
    u16*   WPb = PbT + eS;

    const dim3 blk(256);

    // 0) activations -> bf16 transposed [N,C]
    {
        const dim3 g(N / 64, C / 64, Bn);
        transpose_f32_bf16<<<g, blk, 0, stream>>>(query,   XqT, C, N);
        transpose_f32_bf16<<<g, blk, 0, stream>>>(context, XcT, C, N);
    }
    // 1) projections (A = fp32 weights, B = bf16 XT via TDM)
    {
        const dim3 g(N / 64, F / 256, Bn);
        gemm_bt_wmma<float, u16, false, true><<<g, blk, 0, stream>>>(
            Wq, XqT, Qb,  bq, F, N, C, 0, (long long)N * C, (long long)F * N);
        gemm_bt_wmma<float, u16, false, true><<<g, blk, 0, stream>>>(
            Wk, XcT, Kb,  bk, F, N, C, 0, (long long)N * C, (long long)F * N);
        gemm_bt_wmma<float, u16, true,  true><<<g, blk, 0, stream>>>(
            Wv, XcT, VbT, bv, F, N, C, 0, (long long)N * C, (long long)N * F);
    }
    // 2) S = Q K^T (both operands TDM-staged bf16, K = 3136)
    {
        const dim3 g(F / 64, F / 256, Bn);
        gemm_bt_wmma<u16, float, false, false><<<g, blk, 0, stream>>>(
            Qb, Kb, S, nullptr, F, F, N,
            (long long)F * N, (long long)F * N, (long long)F * F);
    }
    // 3) P^T = softmax(S)^T  (bf16)
    softmax512_to_colmajor<<<dim3(F / 32, 1, Bn), blk, 0, stream>>>(S, PbT);

    // 4) WP = Wo x P  (B = P^T rows)
    {
        const dim3 g(F / 64, F / 256, Bn);
        gemm_bt_wmma<float, u16, false, false><<<g, blk, 0, stream>>>(
            Wo, PbT, WPb, nullptr, F, F, F,
            0, (long long)F * F, (long long)F * F);
    }
    // 5) Y = WP x V + bo  (B = V^T rows) -> f32 d_out [B,512,N]
    {
        const dim3 g(N / 64, F / 256, Bn);
        gemm_bt_wmma<u16, float, false, true><<<g, blk, 0, stream>>>(
            WPb, VbT, (float*)d_out, bo, F, N, F,
            (long long)F * F, (long long)N * F, (long long)F * N);
    }
}
